// MultiHeadAttention_22222160789986
// MI455X (gfx1250) — matrix operand order repaired and checked by reading, not compiled, not run on hardware
//
#include <hip/hip_runtime.h>

typedef __bf16 bhalf;
typedef __attribute__((ext_vector_type(16))) __bf16 v16bf;
typedef __attribute__((ext_vector_type(8)))  __bf16 v8bf;
typedef __attribute__((ext_vector_type(8)))  float  v8f;

#define NB   2
#define SEQ  2048
#define DIM  1024
#define NH   16
#define DKH  64

__device__ __forceinline__ v8f v8f_zero() {
  v8f z;
#pragma unroll
  for (int i = 0; i < 8; ++i) z[i] = 0.0f;
  return z;
}

// A-fragment (16x32 bf16). Per ISA: lanes 0-15 hold rows M=lane, K={0..7,16..23};
// lanes 16-31 hold rows M=lane-16, K={8..15,24..31}.
__device__ __forceinline__ v16bf load_fragA(const bhalf* base, int ld) {
  const int lane = threadIdx.x & 31;
  const bhalf* p = base + (lane & 15) * ld + ((lane >> 4) << 3);
  v8bf lo = *(const v8bf*)(p);
  v8bf hi = *(const v8bf*)(p + 16);
  return __builtin_shufflevector(lo, hi, 0,1,2,3,4,5,6,7,8,9,10,11,12,13,14,15);
}

// B-fragment (32x16 bf16) from a transposed tile Bt[n][k] with row stride ld.
// B layout: lanes 0-15 hold K={0..7,16..23}, lanes 16-31 hold K={8..15,24..31}, col N = lane&15.
__device__ __forceinline__ v16bf load_fragB(const bhalf* base, int ld) {
  const int lane = threadIdx.x & 31;
  const bhalf* p = base + (lane & 15) * ld + ((lane >> 4) << 3);
  v8bf lo = *(const v8bf*)(p);
  v8bf hi = *(const v8bf*)(p + 16);
  return __builtin_shufflevector(lo, hi, 0,1,2,3,4,5,6,7,8,9,10,11,12,13,14,15);
}

__device__ __forceinline__ v8f wmma_bf16(v16bf a, v16bf b, v8f c) {
  return __builtin_amdgcn_wmma_f32_16x16x32_bf16(false, a, false, b, (short)0, c,
                                                 false, false);
}

// Async copy of a 64x64 bf16 tile (row stride `gstride` elems in global, 72 in LDS)
// using GLOBAL_LOAD_ASYNC_TO_LDS_B128: 512 x 16B transfers, 2 per thread (ASYNCcnt).
__device__ __forceinline__ void async_copy_tile64(const bhalf* gbase, int gstride,
                                                  bhalf* lds_base) {
  const int t = threadIdx.x;
#pragma unroll
  for (int i = 0; i < 2; ++i) {
    int id = i * 256 + t;        // 0..511
    int r = id >> 3;             // row 0..63
    int c = (id & 7) * 8;        // col offset 0,8,..,56 (16B chunks)
    unsigned lds = (unsigned)(uintptr_t)(lds_base + r * 72 + c);
    unsigned long long g =
        (unsigned long long)(uintptr_t)(gbase + (size_t)r * gstride + c);
    asm volatile("global_load_async_to_lds_b128 %0, %1, off"
                 :: "v"(lds), "v"(g)
                 : "memory");
  }
}

__device__ __forceinline__ void wait_async0() {
  asm volatile("s_wait_asynccnt 0x0" ::: "memory");
}

// C[4096][1024] = A[4096][1024] x B[1024][1024], bf16 WMMA, fp32 accumulate.
// OUTMODE 0: fp32 row-major. 1: bf16 [b][h][seq][dk]. 2: bf16 [b][h][dk][seq] (V^T).
template <int OUTMODE, typename TA>
__global__ __launch_bounds__(256) void gemm_kernel(const TA* __restrict__ A,
                                                   const float* __restrict__ B,
                                                   void* __restrict__ Cout) {
  constexpr int Kd = DIM, Nn = DIM;
  constexpr int LDS = 40;  // padded stride (32 + 8) -> conflict-free b128 frag loads
  __shared__ __align__(16) bhalf As[128 * LDS];
  __shared__ __align__(16) bhalf Bs[128 * LDS];

  const int n0 = blockIdx.x * 128;
  const int m0 = blockIdx.y * 128;
  const int t = threadIdx.x;
  const int w = t >> 5;
  const int wm = w >> 2;  // 0..1 : 64 rows per wave
  const int wn = w & 3;   // 0..3 : 32 cols per wave

  v8f acc[4][2];
#pragma unroll
  for (int i = 0; i < 4; ++i)
#pragma unroll
    for (int j = 0; j < 2; ++j) acc[i][j] = v8f_zero();

  for (int k0 = 0; k0 < Kd; k0 += 32) {
    __syncthreads();
#pragma unroll
    for (int i = 0; i < 16; ++i) {  // A tile: 128x32, coalesced rows
      int idx = i * 256 + t;
      int r = idx >> 5, c = idx & 31;
      As[r * LDS + c] = (bhalf)A[(m0 + r) * Kd + (k0 + c)];
    }
#pragma unroll
    for (int i = 0; i < 16; ++i) {  // B tile: 32x128, stored transposed Bt[n][k]
      int idx = i * 256 + t;
      int kk = idx >> 7, nn = idx & 127;
      Bs[nn * LDS + kk] = (bhalf)B[(k0 + kk) * Nn + (n0 + nn)];
    }
    __syncthreads();

    v16bf bf[2];
#pragma unroll
    for (int j = 0; j < 2; ++j)
      bf[j] = load_fragB(Bs + (wn * 32 + j * 16) * LDS, LDS);
#pragma unroll
    for (int i = 0; i < 4; ++i) {
      v16bf af = load_fragA(As + (wm * 64 + i * 16) * LDS, LDS);
#pragma unroll
      for (int j = 0; j < 2; ++j) acc[i][j] = wmma_bf16(af, bf[j], acc[i][j]);
    }
  }

  const int lane = t & 31;
  const int hi = (lane >> 4) << 3;  // row offset 0 / 8 per half-wave
  const int cn = lane & 15;
#pragma unroll
  for (int i = 0; i < 4; ++i)
#pragma unroll
    for (int j = 0; j < 2; ++j)
#pragma unroll
      for (int r = 0; r < 8; ++r) {
        int row = m0 + wm * 64 + i * 16 + r + hi;
        int col = n0 + wn * 32 + j * 16 + cn;
        float v = acc[i][j][r];
        if (OUTMODE == 1) {
          int b = row >> 11, s = row & (SEQ - 1);
          int h = col >> 6, dk = col & (DKH - 1);
          ((bhalf*)Cout)[(((size_t)(b * NH + h) * SEQ) + s) * DKH + dk] = (bhalf)v;
        } else if (OUTMODE == 2) {
          int b = row >> 11, s = row & (SEQ - 1);
          int h = col >> 6, dk = col & (DKH - 1);
          ((bhalf*)Cout)[(((size_t)(b * NH + h) * DKH) + dk) * SEQ + s] = (bhalf)v;
        } else {
          ((float*)Cout)[(size_t)row * Nn + col] = v;
        }
      }
}

// Flash-style attention with zero-fill masking (masked scores -> 0, still in softmax).
// Block: 256 threads = 8 waves; each wave owns 16 query rows; block = 128 rows of (b,h).
// K is [b][h][seq][dk]; V is pre-transposed [b][h][dk][seq] so both K and V^T chunk
// tiles are staged into LDS with double-buffered GLOBAL_LOAD_ASYNC_TO_LDS_B128.
__global__ __launch_bounds__(256) void attn_kernel(const bhalf* __restrict__ Q,
                                                   const bhalf* __restrict__ K,
                                                   const bhalf* __restrict__ V,
                                                   const int* __restrict__ mask,
                                                   bhalf* __restrict__ O) {
  constexpr int LDK = 72;  // padded stride (64 + 8)
  __shared__ __align__(16) bhalf KsB[2][64 * LDK];  // K chunk  [key][dk]
  __shared__ __align__(16) bhalf VtB[2][64 * LDK];  // V chunk  [dk][key]
  __shared__ __align__(16) bhalf Ps[128 * LDK];     // probs, C-layout -> A-layout relay

  const int b = blockIdx.z, h = blockIdx.y;
  const int q0 = blockIdx.x * 128;
  const int t = threadIdx.x, w = t >> 5, lane = t & 31;
  const int hi = (lane >> 4) << 3;
  const int cn = lane & 15;

  const bhalf* Qg = Q + (size_t)((b * NH + h) * SEQ) * DKH;
  const bhalf* Kg = K + (size_t)((b * NH + h) * SEQ) * DKH;
  const bhalf* VgT = V + (size_t)((b * NH + h) * DKH) * SEQ;
  const int qrow0 = q0 + w * 16;

  // kick off chunk 0 async staging, then load Q fragments (overlapped)
  async_copy_tile64(Kg, DKH, &KsB[0][0]);
  async_copy_tile64(VgT, SEQ, &VtB[0][0]);

  v16bf qa[2];
#pragma unroll
  for (int kk = 0; kk < 2; ++kk)
    qa[kk] = load_fragA(Qg + (size_t)qrow0 * DKH + kk * 32, DKH);

  float m_run[8], l_run[8];
  v8f oacc[4];
#pragma unroll
  for (int j = 0; j < 8; ++j) { m_run[j] = -1e30f; l_run[j] = 0.0f; }
#pragma unroll
  for (int tt = 0; tt < 4; ++tt) oacc[tt] = v8f_zero();

  constexpr int NC = SEQ / 64;
  int buf = 0;
  for (int ch = 0; ch < NC; ++ch) {
    const int c0 = ch * 64;
    wait_async0();       // this wave's chunk-ch tiles are in LDS
    __syncthreads();     // all waves' tiles visible; all prior back-buffer reads done

    if (ch + 1 < NC) {   // prefetch next chunk into the other buffer (safe: post-barrier)
      const int c1 = c0 + 64;
      async_copy_tile64(Kg + (size_t)c1 * DKH, DKH, &KsB[buf ^ 1][0]);
      async_copy_tile64(VgT + c1, SEQ, &VtB[buf ^ 1][0]);
    }
    const bhalf* Ks = &KsB[buf][0];
    const bhalf* Vt = &VtB[buf][0];

    // S = (Q Kt) / 8   (16 x 64 per wave, 8 WMMA)
    v8f sacc[4];
#pragma unroll
    for (int tt = 0; tt < 4; ++tt) {
      sacc[tt] = v8f_zero();
#pragma unroll
      for (int kk = 0; kk < 2; ++kk) {
        v16bf kbf = load_fragB(Ks + (tt * 16) * LDK + kk * 32, LDK);
        sacc[tt] = wmma_bf16(qa[kk], kbf, sacc[tt]);
      }
    }

    // zero-fill mask + online softmax (rows j+hi per half-wave)
    float pm[4][8];
#pragma unroll
    for (int j = 0; j < 8; ++j) {
      int qr = q0 + w * 16 + j + hi;
      float sv[4];
      float rowmax = -1e30f;
#pragma unroll
      for (int tt = 0; tt < 4; ++tt) {
        int kc = c0 + tt * 16 + cn;
        int mv = mask[(size_t)(b * SEQ + qr) * SEQ + kc];
        float s = sacc[tt][j] * 0.125f;
        s = (mv == 0) ? 0.0f : s;
        sv[tt] = s;
        rowmax = fmaxf(rowmax, s);
      }
#pragma unroll
      for (int off = 8; off >= 1; off >>= 1)
        rowmax = fmaxf(rowmax, __shfl_xor(rowmax, off, 32));
      float mnew = fmaxf(m_run[j], rowmax);
      float resc = __expf(m_run[j] - mnew);
      float psum = 0.0f;
#pragma unroll
      for (int tt = 0; tt < 4; ++tt) {
        float p = __expf(sv[tt] - mnew);
        pm[tt][j] = p;
        psum += p;
      }
#pragma unroll
      for (int off = 8; off >= 1; off >>= 1) psum += __shfl_xor(psum, off, 32);
      m_run[j] = mnew;
      l_run[j] = l_run[j] * resc + psum;
#pragma unroll
      for (int tt = 0; tt < 4; ++tt) oacc[tt][j] = oacc[tt][j] * resc;
    }

    // relay P through LDS to get A-fragment layout (wave-private rows)
#pragma unroll
    for (int j = 0; j < 8; ++j)
#pragma unroll
      for (int tt = 0; tt < 4; ++tt)
        Ps[(w * 16 + j + hi) * LDK + tt * 16 + cn] = (bhalf)pm[tt][j];
    __syncthreads();

    // O += P (16x64) x V (64x64): 8 WMMA per wave
#pragma unroll
    for (int kk = 0; kk < 2; ++kk) {
      v16bf pa = load_fragA(Ps + (w * 16) * LDK + kk * 32, LDK);
#pragma unroll
      for (int tt = 0; tt < 4; ++tt) {
        v16bf vbf = load_fragB(Vt + (tt * 16) * LDK + kk * 32, LDK);
        oacc[tt] = wmma_bf16(pa, vbf, oacc[tt]);
      }
    }
    buf ^= 1;
  }

  // normalize and store O as bf16 in [b][seq][h][dk] (row-major [4096][1024])
#pragma unroll
  for (int tt = 0; tt < 4; ++tt)
#pragma unroll
    for (int j = 0; j < 8; ++j) {
      int qr = q0 + w * 16 + j + hi;
      int dk = tt * 16 + cn;
      float o = oacc[tt][j] / l_run[j];
      O[(size_t)((b * SEQ + qr) * NH + h) * DKH + dk] = (bhalf)o;
    }
}

extern "C" void kernel_launch(void* const* d_in, const int* in_sizes, int n_in,
                              void* d_out, int out_size, void* d_ws, size_t ws_size,
                              hipStream_t stream) {
  const float* Xq = (const float*)d_in[0];
  const float* Xk = (const float*)d_in[1];
  const float* Xv = (const float*)d_in[2];
  const float* Wq = (const float*)d_in[3];
  const float* Wk = (const float*)d_in[4];
  const float* Wv = (const float*)d_in[5];
  const float* Wo = (const float*)d_in[6];
  const int* mask = (const int*)d_in[7];

  const size_t per = (size_t)NB * NH * SEQ * DKH;  // 4.19M bf16 elems = 8 MB
  bhalf* Qb = (bhalf*)d_ws;
  bhalf* Kb = Qb + per;
  bhalf* Vb = Kb + per;  // transposed per-head: [b][h][dk][seq]
  bhalf* Ob = Vb + per;  // attention output, [b][seq][h][dk] == [4096][1024]

  dim3 blk(256);
  dim3 gg(DIM / 128, (NB * SEQ) / 128);

  gemm_kernel<1, float><<<gg, blk, 0, stream>>>(Xq, Wq, (void*)Qb);
  gemm_kernel<1, float><<<gg, blk, 0, stream>>>(Xk, Wk, (void*)Kb);
  gemm_kernel<2, float><<<gg, blk, 0, stream>>>(Xv, Wv, (void*)Vb);

  attn_kernel<<<dim3(SEQ / 128, NH, NB), blk, 0, stream>>>(Qb, Kb, Vb, mask, Ob);

  gemm_kernel<0, bhalf><<<gg, blk, 0, stream>>>(Ob, Wo, d_out);
}
